// HopfieldRetrievalModel_3307124818038
// MI455X (gfx1250) — matrix operand order repaired and checked by reading, not compiled, not run on hardware
//
#include <hip/hip_runtime.h>

// ---------------------------------------------------------------------------
// Hopfield retrieval on MI455X (gfx1250), wave32 + WMMA bf16 + async->LDS.
//
//   xi0 = softmax(B * trg @ memT); 3x { q = xi@mem; xi = softmax(B*q@memT) }
//   out = xi@mem;  pair = l2norm(out) @ l2norm(mem)^T
//
// == 4 applications of A(q) = softmax(B*q@K^T)@K, then a cosine GEMM.
// Softmax normalization is deferred: store unnormalized exp (bf16) + row sums,
// scale by 1/rowsum inside the xi@K GEMM epilogue. Max-subtraction is skipped:
// beta=0.125 and ~N(0,1024) dots give logits with |x| <~ 20, safe in f32 exp.
//
// Tiling: 256 threads = 8 waves (2 M x 4 N), wave tile 64x64 (4x4 WMMA accs),
// block tile 128x256, k-chunk 32. Operands are staged global->LDS with
// GLOBAL_LOAD_ASYNC_TO_LDS_B128 (ASYNCcnt, double buffered, one barrier per
// chunk, strength-reduced per-thread segment descriptors), then all 8
// fragments are ds_load'ed as one clause feeding 16 back-to-back WMMAs.
// ---------------------------------------------------------------------------

#define BETA   0.125f
#define M_Q    4096
#define N_MEM  8192
#define D_DIM  1024

#define KC     32                 // k-chunk (one WMMA k-step)
#define LDA    40                 // padded LDS row pitch in bf16 (80 B)

typedef __attribute__((ext_vector_type(16))) __bf16 v16bf;
typedef __attribute__((ext_vector_type(8)))  __bf16 v8bf;
typedef __attribute__((ext_vector_type(8)))  float  v8f;

// A-fragment (16x32 bf16, ISA 7.12.2): lane l (m = l&15, g = l>>4) holds
// k = {g*8..g*8+7} in VGPR0-3 and k = {16+g*8..16+g*8+7} in VGPR4-7.
// B-fragment (32x16 bf16): lane l (n = l&15, g = l>>4) holds k = {g*8..g*8+7} then {16+g*8..16+g*8+7}.
__device__ __forceinline__ v16bf make_frag(v8bf lo, v8bf hi) {
    return __builtin_shufflevector(lo, hi, 0,1,2,3,4,5,6,7,8,9,10,11,12,13,14,15);
}

// Async 16B global->LDS copy (GLOBAL_LOAD_ASYNC_TO_LDS_B128, ASYNCcnt).
__device__ __forceinline__ void async_copy16(unsigned lds_addr, const void* gaddr) {
    asm volatile("global_load_async_to_lds_b128 %0, %1, off"
                 :: "v"(lds_addr), "v"(gaddr) : "memory");
}
__device__ __forceinline__ void wait_async0() {
    asm volatile("s_wait_asynccnt 0x0" ::: "memory");
}
__device__ __forceinline__ unsigned lds_addr_of(const void* p) {
    // LDS flat aperture: addr[31:0] is the LDS byte offset (ISA 10.2).
    return (unsigned)(unsigned long long)p;
}

template<int MODE>  // 0 = scores->exp(xi), 1 = xi@K -> q, 2 = cosine pair
__global__ void __launch_bounds__(256)
hopfield_gemm_nt(const __bf16* __restrict__ A,   // [M x Kd] row-major
                 const __bf16* __restrict__ B,   // [Nb x Kd] row-major
                 int Kd, int Nb,
                 __bf16* __restrict__ outB,      // MODE 0/1 bf16 output [M x Nb]
                 float*  __restrict__ outF,      // MODE 2 f32 output   [M x Nb]
                 const float* __restrict__ rowScaleIn, // MODE1: rowsum; MODE2: rq_inv
                 float*  __restrict__ rowPartOut,      // MODE0/1: per-block row partials
                 const float* __restrict__ colScaleIn) // MODE2: rk_inv
{
    __shared__ __bf16 As[2][128][LDA];   // 20 KB
    __shared__ __bf16 Bs[2][256][LDA];   // 40 KB
    __shared__ float  sred[4 * 128];     //  2 KB

    const int lane = threadIdx.x & 31;
    const int wave = threadIdx.x >> 5;
    const int wm = wave >> 2;            // 0..1  (M direction)
    const int wn = wave & 3;             // 0..3  (N direction)
    const int g  = lane >> 4;            // lane half
    const int l16 = lane & 15;

    const int m_blk  = blockIdx.y * 128;
    const int n_blk  = blockIdx.x * 256;
    const int m_base = m_blk + wm * 64;
    const int n_base = n_blk + wn * 64;

    v8f acc[4][4];
#pragma unroll
    for (int i = 0; i < 4; ++i)
#pragma unroll
        for (int j = 0; j < 4; ++j)
#pragma unroll
            for (int r = 0; r < 8; ++r) acc[i][j][r] = 0.f;

    // --- per-thread staging descriptors (6 x 16B segments per thread/chunk) ---
    const unsigned ATILE = 128 * LDA * 2;     // bytes per A buffer
    const unsigned BTILE = 256 * LDA * 2;     // bytes per B buffer
    const char* gcur[6];                      // global source, advanced 64 B/chunk
    unsigned    lds0[6];                      // LDS dest within buffer 0
    {
        const int t = threadIdx.x;
#pragma unroll
        for (int p = 0; p < 2; ++p) {         // A: 512 segments (128 rows x 64 B)
            const int s = t + p * 256, row = s >> 2, cb = (s & 3) * 16;
            gcur[p] = (const char*)(A + (size_t)(m_blk + row) * Kd) + cb;
            lds0[p] = lds_addr_of(&As[0][0][0] + row * LDA) + cb;
        }
#pragma unroll
        for (int p = 0; p < 4; ++p) {         // B: 1024 segments (256 rows x 64 B)
            const int s = t + p * 256, row = s >> 2, cb = (s & 3) * 16;
            gcur[2 + p] = (const char*)(B + (size_t)(n_blk + row) * Kd) + cb;
            lds0[2 + p] = lds_addr_of(&Bs[0][0][0] + row * LDA) + cb;
        }
    }

    // prologue: stage chunk 0 into buffer 0
#pragma unroll
    for (int p = 0; p < 6; ++p) {
        async_copy16(lds0[p], gcur[p]);
        gcur[p] += KC * 2;
    }

    int buf = 0;
    for (int kc = 0; kc < Kd; kc += KC) {
        wait_async0();        // my async stores for buffer `buf` are in LDS
        __syncthreads();      // everyone staged buf; everyone finished computing buf^1
        if (kc + KC < Kd) {
            const unsigned bsel = (unsigned)(buf ^ 1);
#pragma unroll
            for (int p = 0; p < 6; ++p) {
                const unsigned tile = (p < 2) ? ATILE : BTILE;
                async_copy16(lds0[p] + bsel * tile, gcur[p]);
                gcur[p] += KC * 2;
            }
        }

        const __bf16* AsB = &As[buf][0][0];
        const __bf16* BsB = &Bs[buf][0][0];

        v16bf a[4], b[4];
#pragma unroll
        for (int i = 0; i < 4; ++i) {
            const __bf16* pr = AsB + (wm * 64 + i * 16 + l16) * LDA + g * 8;
            a[i] = make_frag(*(const v8bf*)pr, *(const v8bf*)(pr + 16));
        }
#pragma unroll
        for (int j = 0; j < 4; ++j) {
            const __bf16* pr = BsB + (wn * 64 + j * 16 + l16) * LDA + g * 8;
            b[j] = make_frag(*(const v8bf*)pr, *(const v8bf*)(pr + 16));
        }
#pragma unroll
        for (int j = 0; j < 4; ++j)
#pragma unroll
            for (int i = 0; i < 4; ++i)
                acc[i][j] = __builtin_amdgcn_wmma_f32_16x16x32_bf16(
                    false, a[i], false, b[j], (short)0, acc[i][j], false, false);
        buf ^= 1;
    }
    __syncthreads();

    // C layout (16x16 f32): VGPR r, lanes 0-15 -> M=r (n=lane), lanes 16-31 -> M=8+r.
    // => element (i,j,r): m = m_base + i*16 + g*8 + r ; n = n_base + j*16 + l16.

    if constexpr (MODE == 0) {
        // exp(beta * S); row sums reduced deterministically (lanes -> LDS -> fixed order).
        float rs[4][8];
#pragma unroll
        for (int i = 0; i < 4; ++i)
#pragma unroll
            for (int r = 0; r < 8; ++r) rs[i][r] = 0.f;
#pragma unroll
        for (int i = 0; i < 4; ++i)
#pragma unroll
            for (int j = 0; j < 4; ++j)
#pragma unroll
                for (int r = 0; r < 8; ++r) {
                    float e = __expf(acc[i][j][r] * BETA);
                    acc[i][j][r] = e;
                    rs[i][r] += e;
                }
#pragma unroll
        for (int i = 0; i < 4; ++i)
#pragma unroll
            for (int r = 0; r < 8; ++r) {
                float v = rs[i][r];
                v += __shfl_xor(v, 1); v += __shfl_xor(v, 2);
                v += __shfl_xor(v, 4); v += __shfl_xor(v, 8);
                rs[i][r] = v;   // lane l16==0 of each half holds the 16-wide sum
            }
        if (l16 == 0) {
#pragma unroll
            for (int i = 0; i < 4; ++i)
#pragma unroll
                for (int r = 0; r < 8; ++r)
                    sred[wn * 128 + wm * 64 + i * 16 + g * 8 + r] = rs[i][r];
        }
        __syncthreads();
        if (threadIdx.x < 128) {
            float s = sred[threadIdx.x] + sred[128 + threadIdx.x]
                    + sred[256 + threadIdx.x] + sred[384 + threadIdx.x];
            rowPartOut[(size_t)(blockIdx.y * 128 + threadIdx.x) * gridDim.x + blockIdx.x] = s;
        }
        // store unnormalized exp as bf16 xi
#pragma unroll
        for (int i = 0; i < 4; ++i)
#pragma unroll
            for (int r = 0; r < 8; ++r) {
                const int m = m_base + i * 16 + g * 8 + r;
                __bf16* prow = outB + (size_t)m * Nb + n_base + l16;
#pragma unroll
                for (int j = 0; j < 4; ++j) prow[j * 16] = (__bf16)acc[i][j][r];
            }
    } else if constexpr (MODE == 1) {
        // q = (xi @ K) / rowsum ; also per-row sum(q^2) partials for the final l2norm.
        float inv[4][8], ss[4][8];
#pragma unroll
        for (int i = 0; i < 4; ++i)
#pragma unroll
            for (int r = 0; r < 8; ++r) {
                inv[i][r] = 1.0f / rowScaleIn[m_base + i * 16 + g * 8 + r];
                ss[i][r] = 0.f;
            }
#pragma unroll
        for (int i = 0; i < 4; ++i)
#pragma unroll
            for (int r = 0; r < 8; ++r) {
                const int m = m_base + i * 16 + g * 8 + r;
                __bf16* prow = outB + (size_t)m * Nb + n_base + l16;
#pragma unroll
                for (int j = 0; j < 4; ++j) {
                    float q = acc[i][j][r] * inv[i][r];
                    ss[i][r] += q * q;
                    prow[j * 16] = (__bf16)q;
                }
            }
#pragma unroll
        for (int i = 0; i < 4; ++i)
#pragma unroll
            for (int r = 0; r < 8; ++r) {
                float v = ss[i][r];
                v += __shfl_xor(v, 1); v += __shfl_xor(v, 2);
                v += __shfl_xor(v, 4); v += __shfl_xor(v, 8);
                ss[i][r] = v;
            }
        if (l16 == 0) {
#pragma unroll
            for (int i = 0; i < 4; ++i)
#pragma unroll
                for (int r = 0; r < 8; ++r)
                    sred[wn * 128 + wm * 64 + i * 16 + g * 8 + r] = ss[i][r];
        }
        __syncthreads();
        if (threadIdx.x < 128) {
            float s = sred[threadIdx.x] + sred[128 + threadIdx.x]
                    + sred[256 + threadIdx.x] + sred[384 + threadIdx.x];
            rowPartOut[(size_t)(blockIdx.y * 128 + threadIdx.x) * gridDim.x + blockIdx.x] = s;
        }
    } else {
        // pair[m][n] = (out_m . mem_n) * rq_inv[m] * rk_inv[n]  (f32)
        float rq[4][8];
#pragma unroll
        for (int i = 0; i < 4; ++i)
#pragma unroll
            for (int r = 0; r < 8; ++r)
                rq[i][r] = rowScaleIn[m_base + i * 16 + g * 8 + r];
        float cs[4];
#pragma unroll
        for (int j = 0; j < 4; ++j) cs[j] = colScaleIn[n_base + j * 16 + l16];
#pragma unroll
        for (int i = 0; i < 4; ++i)
#pragma unroll
            for (int r = 0; r < 8; ++r) {
                const int m = m_base + i * 16 + g * 8 + r;
                float* prow = outF + (size_t)m * Nb + n_base + l16;
#pragma unroll
                for (int j = 0; j < 4; ++j)
                    prow[j * 16] = acc[i][j][r] * rq[i][r] * cs[j];
            }
    }
}

// f32 -> bf16 of memory (plus transposed copy) and trg.
__global__ void prep_convert(const float* __restrict__ mem, const float* __restrict__ trg,
                             __bf16* __restrict__ Kbf, __bf16* __restrict__ KT,
                             __bf16* __restrict__ Q0)
{
    const size_t NT = (size_t)N_MEM * D_DIM;
    const size_t MT = (size_t)M_Q * D_DIM;
    const size_t stride = (size_t)gridDim.x * blockDim.x;
    for (size_t i = (size_t)blockIdx.x * blockDim.x + threadIdx.x; i < NT; i += stride) {
        __bf16 b = (__bf16)mem[i];
        Kbf[i] = b;
        size_t n = i / D_DIM, d = i - n * D_DIM;
        KT[d * (size_t)N_MEM + n] = b;
    }
    for (size_t i = (size_t)blockIdx.x * blockDim.x + threadIdx.x; i < MT; i += stride)
        Q0[i] = (__bf16)trg[i];
}

// 1/max(||mem_row||, eps), one wave per row (f32 source for accuracy).
__global__ void rownorm_mem(const float* __restrict__ mem, float* __restrict__ rk_inv)
{
    const int row  = (blockIdx.x * blockDim.x + threadIdx.x) >> 5;
    const int lane = threadIdx.x & 31;
    if (row >= N_MEM) return;
    const float* p = mem + (size_t)row * D_DIM;
    float s = 0.f;
    for (int t = lane; t < D_DIM; t += 32) { float v = p[t]; s += v * v; }
    s += __shfl_xor(s, 1);  s += __shfl_xor(s, 2);  s += __shfl_xor(s, 4);
    s += __shfl_xor(s, 8);  s += __shfl_xor(s, 16);
    if (lane == 0) rk_inv[row] = 1.0f / fmaxf(sqrtf(s), 1e-12f);
}

// Deterministic fixed-order reduction of per-block partials.
__global__ void reduce_rows(const float* __restrict__ part, int P,
                            float* __restrict__ outv, int n)
{
    int i = blockIdx.x * blockDim.x + threadIdx.x;
    if (i < n) {
        float s = 0.f;
        for (int p = 0; p < P; ++p) s += part[(size_t)i * P + p];
        outv[i] = s;
    }
}

__global__ void finalize_rq(const float* __restrict__ part, float* __restrict__ rq_inv)
{
    int i = blockIdx.x * blockDim.x + threadIdx.x;
    if (i < M_Q) {
        float s = part[i * 4 + 0] + part[i * 4 + 1] + part[i * 4 + 2] + part[i * 4 + 3];
        rq_inv[i] = 1.0f / fmaxf(sqrtf(s), 1e-12f);
    }
}

extern "C" void kernel_launch(void* const* d_in, const int* in_sizes, int n_in,
                              void* d_out, int out_size, void* d_ws, size_t ws_size,
                              hipStream_t stream)
{
    (void)in_sizes; (void)n_in; (void)out_size; (void)ws_size;
    const float* mem = (const float*)d_in[0];   // [8192 x 1024] f32
    const float* trg = (const float*)d_in[1];   // [4096 x 1024] f32
    float* out = (float*)d_out;                 // [4096 x 8192] f32

    char* ws = (char*)d_ws;
    __bf16* Kbf = (__bf16*)ws;  ws += (size_t)N_MEM * D_DIM * 2;   // 16 MB
    __bf16* KT  = (__bf16*)ws;  ws += (size_t)D_DIM * N_MEM * 2;   // 16 MB
    __bf16* Qa  = (__bf16*)ws;  ws += (size_t)M_Q * D_DIM * 2;     //  8 MB
    __bf16* Qb  = (__bf16*)ws;  ws += (size_t)M_Q * D_DIM * 2;     //  8 MB
    __bf16* xi  = (__bf16*)ws;  ws += (size_t)M_Q * N_MEM * 2;     // 64 MB
    float* rowPart = (float*)ws; ws += (size_t)M_Q * 32 * 4;       // 512 KB
    float* rowsum  = (float*)ws; ws += (size_t)M_Q * 4;
    float* ssPart  = (float*)ws; ws += (size_t)M_Q * 4 * 4;
    float* rq_inv  = (float*)ws; ws += (size_t)M_Q * 4;
    float* rk_inv  = (float*)ws; ws += (size_t)N_MEM * 4;

    prep_convert<<<4096, 256, 0, stream>>>(mem, trg, Kbf, KT, Qa);
    rownorm_mem<<<(N_MEM * 32) / 256, 256, 0, stream>>>(mem, rk_inv);

    dim3 gScores(N_MEM / 256, M_Q / 128);   // (32, 32)
    dim3 gGemm2 (D_DIM / 256, M_Q / 128);   // ( 4, 32)

    __bf16* q_in = Qa;
    __bf16* q_out = Qb;
    for (int t = 0; t < 4; ++t) {
        // xi = exp(beta * q K^T), rowPart = per-block row sums
        hopfield_gemm_nt<0><<<gScores, 256, 0, stream>>>(
            q_in, Kbf, D_DIM, N_MEM, xi, nullptr, nullptr, rowPart, nullptr);
        reduce_rows<<<16, 256, 0, stream>>>(rowPart, 32, rowsum, M_Q);
        // q_out = (xi @ mem) / rowsum, ssPart = per-block sum(q^2)
        hopfield_gemm_nt<1><<<gGemm2, 256, 0, stream>>>(
            xi, KT, N_MEM, D_DIM, q_out, nullptr, rowsum, ssPart, nullptr);
        __bf16* tswap = q_in; q_in = q_out; q_out = tswap;
    }

    finalize_rq<<<16, 256, 0, stream>>>(ssPart, rq_inv);  // from final step's ssPart
    hopfield_gemm_nt<2><<<gScores, 256, 0, stream>>>(
        q_in, Kbf, D_DIM, N_MEM, nullptr, out, rq_inv, nullptr, rk_inv);
}
